// StarTransformerEncoderLayer_1460288881172
// MI455X (gfx1250) — matrix operand order repaired and checked by reading, not compiled, not run on hardware
//
#include <hip/hip_runtime.h>

#define BB   16
#define LL   2048
#define DD   256
#define HH   8
#define FF   1024
#define NCYC 2
#define DEP  32

typedef __attribute__((ext_vector_type(16))) __bf16 v16bf;
typedef __attribute__((ext_vector_type(8)))  float  v8f;

union FragU { v16bf v; unsigned int u[8]; };

__device__ __forceinline__ unsigned short f2bf(float f) {
  unsigned int u = __builtin_bit_cast(unsigned int, f);
  u += 0x7FFFu + ((u >> 16) & 1u);          // round-to-nearest-even
  return (unsigned short)(u >> 16);
}

// ---- WMMA fragment loaders -------------------------------------------------
// A: 16x32 bf16 chunk starting at rowPtr. Lane m<16 -> row m, K {0..7,16..23};
//    lanes 16..31 -> same rows, K {8..15,24..31}. Pairs packed per dword.
__device__ __forceinline__ v16bf load_a_frag(const unsigned short* rowPtr, int half) {
  FragU f;
  const unsigned int* p0 = (const unsigned int*)(rowPtr + half * 8);
  const unsigned int* p1 = (const unsigned int*)(rowPtr + 16 + half * 8);
#pragma unroll
  for (int t = 0; t < 4; ++t) { f.u[t] = p0[t]; f.u[4 + t] = p1[t]; }
  return f.v;
}
// B: 32x16 bf16 from row-major (K x N) weight. Lane l holds column N=l%16,
// K {0..7,16..23} (lanes<16) or K {8..15,24..31} (lanes>=16), strided by N.
// ptr must already point at (chunk_row0 * N + (lane%16) + n0).
__device__ __forceinline__ v16bf load_b_frag(const unsigned short* ptr, int half, int N) {
  FragU f;
#pragma unroll
  for (int t = 0; t < 4; ++t) {
    int k0 = half * 8 + 2 * t, k1 = 16 + half * 8 + 2 * t;
    f.u[t] = (unsigned int)ptr[(size_t)k0 * N] | ((unsigned int)ptr[(size_t)(k0 + 1) * N] << 16);
    f.u[4 + t] = (unsigned int)ptr[(size_t)k1 * N] | ((unsigned int)ptr[(size_t)(k1 + 1) * N] << 16);
  }
  return f.v;
}

// ---- BF16 GEMM: C(MxN) = A(MxK) * W(KxN), f32 accumulate --------------------
// Compile-time N,K -> all addresses are base+immediate. One wave computes a
// (16*MT)x64 tile (MT*4 WMMA accumulators); register double-buffer overlaps
// VMEM with WMMA, and MT=2 amortizes each B-fragment over 2 WMMAs.
template <int N, int K, int MT, bool BIAS, bool RELU, bool F32OUT, bool BF16OUT>
__global__ void __launch_bounds__(256)
gemm_bf16(const unsigned short* __restrict__ A, const unsigned short* __restrict__ W,
          const float* __restrict__ bias, float* __restrict__ Cf,
          unsigned short* __restrict__ Cb, int M, int totalWaves) {
  int wave = blockIdx.x * 8 + (threadIdx.x >> 5);
  if (wave >= totalWaves) return;
  int lane = threadIdx.x & 31;
  constexpr int ntb = N >> 6;
  int row0 = (wave / ntb) * (16 * MT);
  int col0 = (wave % ntb) << 6;
  int half = lane >> 4;
  int mr   = lane & 15;
  const unsigned short* aptr[MT];
#pragma unroll
  for (int m = 0; m < MT; ++m) aptr[m] = A + (size_t)(row0 + m * 16 + mr) * K;
  const unsigned short* bptr = W + (size_t)mr + col0;

  v8f acc[MT][4];
#pragma unroll
  for (int m = 0; m < MT; ++m)
#pragma unroll
    for (int t = 0; t < 4; ++t)
#pragma unroll
      for (int r = 0; r < 8; ++r) acc[m][t][r] = 0.0f;

  // prologue: chunk 0 in registers
  v16bf a_cur[MT], b_cur[4];
#pragma unroll
  for (int m = 0; m < MT; ++m) a_cur[m] = load_a_frag(aptr[m], half);
#pragma unroll
  for (int t = 0; t < 4; ++t) b_cur[t] = load_b_frag(bptr + t * 16, half, N);

#pragma unroll
  for (int k0 = 32; k0 < K; k0 += 32) {
    // issue next chunk's loads before consuming current registers
    v16bf a_nxt[MT], b_nxt[4];
#pragma unroll
    for (int m = 0; m < MT; ++m) a_nxt[m] = load_a_frag(aptr[m] + k0, half);
#pragma unroll
    for (int t = 0; t < 4; ++t) b_nxt[t] = load_b_frag(bptr + (size_t)k0 * N + t * 16, half, N);
#pragma unroll
    for (int m = 0; m < MT; ++m)
#pragma unroll
      for (int t = 0; t < 4; ++t)
        acc[m][t] = __builtin_amdgcn_wmma_f32_16x16x32_bf16(false, a_cur[m], false, b_cur[t],
                                                            (short)0, acc[m][t], false, false);
#pragma unroll
    for (int m = 0; m < MT; ++m) a_cur[m] = a_nxt[m];
#pragma unroll
    for (int t = 0; t < 4; ++t) b_cur[t] = b_nxt[t];
  }
#pragma unroll
  for (int m = 0; m < MT; ++m)
#pragma unroll
    for (int t = 0; t < 4; ++t)
      acc[m][t] = __builtin_amdgcn_wmma_f32_16x16x32_bf16(false, a_cur[m], false, b_cur[t],
                                                          (short)0, acc[m][t], false, false);

  // C layout: lane n<16 -> col n, VGPR r -> row r (lanes>=16: row r+8)
#pragma unroll
  for (int mt = 0; mt < MT; ++mt)
#pragma unroll
    for (int t = 0; t < 4; ++t) {
      int n = col0 + t * 16 + mr;
      float bv = BIAS ? bias[n] : 0.0f;
#pragma unroll
      for (int r = 0; r < 8; ++r) {
        int m = row0 + mt * 16 + (half << 3) + r;
        float v = acc[mt][t][r] + bv;
        if (RELU) v = v > 0.0f ? v : 0.0f;
        size_t idx = (size_t)m * N + n;
        if (F32OUT)  Cf[idx] = v;
        if (BF16OUT) Cb[idx] = f2bf(v);
      }
    }
}

// ---- Ring attention: per token, 5 keys {h[i+1],h[i],h[i-1],e[i],s} ----------
// One wave per token; 4 lanes per head (8 dims each); shfl-xor dot reduce.
__global__ void __launch_bounds__(256)
ring_attn(const float* __restrict__ Q, const float* __restrict__ Kh,
          const float* __restrict__ Ke, const float* __restrict__ Ks,
          const float* __restrict__ Vh, const float* __restrict__ Ve,
          const float* __restrict__ Vs, unsigned short* __restrict__ outb) {
  int token = blockIdx.x * 8 + (threadIdx.x >> 5);
  if (token >= BB * LL) return;
  int lane = threadIdx.x & 31;
  int b = token / LL, pos = token % LL;
  int dbase = lane * 8;  // == head*32 + sub*8
  size_t iC = (size_t)token * DD;
  size_t iN = ((size_t)b * LL + ((pos + 1) % LL)) * DD;        // roll(h,-1)
  size_t iP = ((size_t)b * LL + ((pos + LL - 1) % LL)) * DD;   // roll(h,+1)
  size_t iS = (size_t)b * DD;
  const float* kr[5] = {Kh + iN, Kh + iC, Kh + iP, Ke + iC, Ks + iS};
  const float* vr[5] = {Vh + iN, Vh + iC, Vh + iP, Ve + iC, Vs + iS};

  float q[8];
#pragma unroll
  for (int i = 0; i < 8; ++i) q[i] = Q[iC + dbase + i];

  const float scale = 0.17677669529663687f;  // 1/sqrt(32)
  float lg[5];
#pragma unroll
  for (int j = 0; j < 5; ++j) {
    float p = 0.0f;
#pragma unroll
    for (int i = 0; i < 8; ++i) p += q[i] * kr[j][dbase + i];
    p += __shfl_xor(p, 1);
    p += __shfl_xor(p, 2);
    lg[j] = p * scale;
  }
  float mx = lg[0];
#pragma unroll
  for (int j = 1; j < 5; ++j) mx = fmaxf(mx, lg[j]);
  float w[5], sum = 0.0f;
#pragma unroll
  for (int j = 0; j < 5; ++j) { w[j] = __expf(lg[j] - mx); sum += w[j]; }
  float inv = 1.0f / sum;
#pragma unroll
  for (int i = 0; i < 8; ++i) {
    float o = 0.0f;
#pragma unroll
    for (int j = 0; j < 5; ++j) o += w[j] * vr[j][dbase + i];
    outb[iC + dbase + i] = f2bf(o * inv);
  }
}

// ---- Relay attention: s attends over [s; h_new] (2049 keys) -----------------
// One 256-thread workgroup per (batch, head). Deterministic LDS reductions.
__global__ void __launch_bounds__(256)
relay_attn(const float* __restrict__ Qs, const float* __restrict__ Ks,
           const float* __restrict__ Kh, const float* __restrict__ Vs,
           const float* __restrict__ Vh, unsigned short* __restrict__ outb) {
  __shared__ float qsh[DEP];
  __shared__ float red[256];
  __shared__ float vtab[256][DEP + 1];
  int b = blockIdx.x >> 3, head = blockIdx.x & 7;
  int tid = threadIdx.x;
  if (tid < DEP) qsh[tid] = Qs[(size_t)b * DD + head * DEP + tid];
  __syncthreads();

  const float scale = 0.17677669529663687f;
  float lg[9];
  float lmax = -3.0e38f;
  int cnt = 0;
  for (int j = tid; j < LL + 1; j += 256) {
    const float* krw = (j == 0) ? (Ks + (size_t)b * DD)
                                : (Kh + ((size_t)b * LL + (j - 1)) * DD);
    float dot = 0.0f;
#pragma unroll 8
    for (int d = 0; d < DEP; ++d) dot += qsh[d] * krw[head * DEP + d];
    lg[cnt] = dot * scale;
    lmax = fmaxf(lmax, lg[cnt]);
    ++cnt;
  }
  red[tid] = lmax; __syncthreads();
  for (int s2 = 128; s2 > 0; s2 >>= 1) {
    if (tid < s2) red[tid] = fmaxf(red[tid], red[tid + s2]);
    __syncthreads();
  }
  float gmax = red[0]; __syncthreads();

  float vacc[DEP];
#pragma unroll
  for (int d = 0; d < DEP; ++d) vacc[d] = 0.0f;
  float lsum = 0.0f;
  cnt = 0;
  for (int j = tid; j < LL + 1; j += 256) {
    float wv = __expf(lg[cnt++] - gmax);
    lsum += wv;
    const float* vrw = (j == 0) ? (Vs + (size_t)b * DD)
                                : (Vh + ((size_t)b * LL + (j - 1)) * DD);
#pragma unroll 8
    for (int d = 0; d < DEP; ++d) vacc[d] += wv * vrw[head * DEP + d];
  }
  red[tid] = lsum; __syncthreads();
  for (int s2 = 128; s2 > 0; s2 >>= 1) {
    if (tid < s2) red[tid] += red[tid + s2];
    __syncthreads();
  }
  float gsum = red[0]; __syncthreads();

  for (int d = 0; d < DEP; ++d) vtab[tid][d] = vacc[d];
  __syncthreads();
  if (tid < DEP) {
    float a = 0.0f;
    for (int t = 0; t < 256; ++t) a += vtab[t][tid];
    outb[(size_t)b * DD + head * DEP + tid] = f2bf(a / gsum);
  }
}

// ---- Fused residual-add + LayerNorm: wave per row ---------------------------
template <bool BF16OUT>
__global__ void __launch_bounds__(256)
add_ln(const float* __restrict__ X, const float* __restrict__ Y,
       const float* __restrict__ g, const float* __restrict__ be,
       float* __restrict__ outf, unsigned short* __restrict__ outb, int rows) {
  int row = blockIdx.x * 8 + (threadIdx.x >> 5);
  if (row >= rows) return;
  int lane = threadIdx.x & 31;
  int d0 = lane * 8;
  size_t base = (size_t)row * DD;
  float v[8];
  float s = 0.0f;
#pragma unroll
  for (int i = 0; i < 8; ++i) { v[i] = X[base + d0 + i] + Y[base + d0 + i]; s += v[i]; }
#pragma unroll
  for (int m = 16; m >= 1; m >>= 1) s += __shfl_xor(s, m);
  float mu = s * (1.0f / DD);
  float var = 0.0f;
#pragma unroll
  for (int i = 0; i < 8; ++i) { float d = v[i] - mu; var += d * d; }
#pragma unroll
  for (int m = 16; m >= 1; m >>= 1) var += __shfl_xor(var, m);
  float inv = rsqrtf(var * (1.0f / DD) + 1e-6f);
#pragma unroll
  for (int i = 0; i < 8; ++i) {
    float o = g[d0 + i] * (v[i] - mu) * inv + be[d0 + i];
    outf[base + d0 + i] = o;
    if (BF16OUT) outb[base + d0 + i] = f2bf(o);
  }
}

// ---- Small helpers ----------------------------------------------------------
__global__ void __launch_bounds__(256)
einit_kernel(const float* __restrict__ e, unsigned short* __restrict__ eb,
             unsigned short* __restrict__ hb, float* __restrict__ hf, int n) {
  int i = blockIdx.x * 256 + threadIdx.x;
  if (i < n) { float v = e[i]; hf[i] = v; unsigned short x = f2bf(v); eb[i] = x; hb[i] = x; }
}

__global__ void __launch_bounds__(256)
f2b_kernel(const float* __restrict__ x, unsigned short* __restrict__ y, int n) {
  int i = blockIdx.x * 256 + threadIdx.x;
  if (i < n) y[i] = f2bf(x[i]);
}

__global__ void __launch_bounds__(256)
copy_kernel(const float* __restrict__ x, float* __restrict__ y, int n) {
  int i = blockIdx.x * 256 + threadIdx.x;
  if (i < n) y[i] = x[i];
}

__global__ void __launch_bounds__(256)
smean_kernel(const float* __restrict__ e, float* __restrict__ sf,
             unsigned short* __restrict__ sb) {
  int b = blockIdx.x, d = threadIdx.x;
  float acc = 0.0f;
  const float* p = e + (size_t)b * LL * DD + d;
  for (int l = 0; l < LL; ++l) acc += p[(size_t)l * DD];
  float m = acc * (1.0f / LL);
  sf[b * DD + d] = m;
  sb[b * DD + d] = f2bf(m);
}

// ---- Host side --------------------------------------------------------------
template <int N, int K, int MT, bool BIAS, bool RELU, bool F32O, bool B16O>
static void run_gemm(const unsigned short* A, const unsigned short* W, const float* bias,
                     float* Cf, unsigned short* Cb, int M, hipStream_t st) {
  int totalWaves = (M / (16 * MT)) * (N / 64);
  int blocks = (totalWaves + 7) / 8;
  gemm_bf16<N, K, MT, BIAS, RELU, F32O, B16O><<<blocks, 256, 0, st>>>(A, W, bias, Cf, Cb, M, totalWaves);
}

extern "C" void kernel_launch(void* const* d_in, const int* in_sizes, int n_in,
                              void* d_out, int out_size, void* d_ws, size_t ws_size,
                              hipStream_t stream) {
  (void)in_sizes; (void)n_in; (void)out_size; (void)ws_size;
  const float* e   = (const float*)d_in[0];
  const float* Wq  = (const float*)d_in[1];
  const float* Wk  = (const float*)d_in[2];
  const float* Wv  = (const float*)d_in[3];
  const float* Wo  = (const float*)d_in[4];
  const float* bo  = (const float*)d_in[5];
  const float* W1  = (const float*)d_in[6];
  const float* b1  = (const float*)d_in[7];
  const float* W2  = (const float*)d_in[8];
  const float* b2  = (const float*)d_in[9];
  const float* g1  = (const float*)d_in[10];
  const float* be1 = (const float*)d_in[11];
  const float* g2  = (const float*)d_in[12];
  const float* be2 = (const float*)d_in[13];

  const long ND = (long)BB * LL * DD;  // 8388608
  char* wp = (char*)d_ws;
  auto alloc = [&](size_t bytes) -> char* {
    char* p = wp;
    wp += (bytes + 255) & ~(size_t)255;
    return p;
  };
  unsigned short* eb    = (unsigned short*)alloc(ND * 2);
  unsigned short* hb    = (unsigned short*)alloc(ND * 2);
  float*          hf    = (float*)alloc(ND * 4);
  float*          Qf    = (float*)alloc(ND * 4);
  float*          Khf   = (float*)alloc(ND * 4);
  float*          Vhf   = (float*)alloc(ND * 4);
  float*          Kef   = (float*)alloc(ND * 4);   // Kef..Vef contiguous: reused as ffn1
  float*          Vef   = (float*)alloc(ND * 4);
  unsigned short* attnb = (unsigned short*)alloc(ND * 2);
  float*          sf    = (float*)alloc((size_t)BB * DD * 4);
  unsigned short* sb    = (unsigned short*)alloc((size_t)BB * DD * 2);
  float*          Ksf   = (float*)alloc((size_t)BB * DD * 4);
  float*          Vsf   = (float*)alloc((size_t)BB * DD * 4);
  float*          Qsf   = (float*)alloc((size_t)BB * DD * 4);
  unsigned short* satb  = (unsigned short*)alloc((size_t)BB * DD * 2);
  unsigned short* Wqb   = (unsigned short*)alloc((size_t)DD * DD * 2);
  unsigned short* Wkb   = (unsigned short*)alloc((size_t)DD * DD * 2);
  unsigned short* Wvb   = (unsigned short*)alloc((size_t)DD * DD * 2);
  unsigned short* Wob   = (unsigned short*)alloc((size_t)DD * DD * 2);
  unsigned short* W1b   = (unsigned short*)alloc((size_t)DD * FF * 2);
  unsigned short* W2b   = (unsigned short*)alloc((size_t)FF * DD * 2);
  // Aliases for post-cycle stages (cycle buffers are dead by then):
  float*          out1f = Khf;
  unsigned short* out1b = attnb;
  unsigned short* ffn1b = (unsigned short*)Kef;  // 67MB == Kef+Vef (contiguous)
  float*          ffn2f = Qf;

  // Weight conversion (layout preserved: row-major K x N feeds B-fragments directly)
  f2b_kernel<<<(DD * DD + 255) / 256, 256, 0, stream>>>(Wq, Wqb, DD * DD);
  f2b_kernel<<<(DD * DD + 255) / 256, 256, 0, stream>>>(Wk, Wkb, DD * DD);
  f2b_kernel<<<(DD * DD + 255) / 256, 256, 0, stream>>>(Wv, Wvb, DD * DD);
  f2b_kernel<<<(DD * DD + 255) / 256, 256, 0, stream>>>(Wo, Wob, DD * DD);
  f2b_kernel<<<(DD * FF + 255) / 256, 256, 0, stream>>>(W1, W1b, DD * FF);
  f2b_kernel<<<(FF * DD + 255) / 256, 256, 0, stream>>>(W2, W2b, FF * DD);

  // h = e ; s = mean_L(e) ; hoisted e-projections
  einit_kernel<<<(int)(ND / 256), 256, 0, stream>>>(e, eb, hb, hf, (int)ND);
  smean_kernel<<<BB, DD, 0, stream>>>(e, sf, sb);
  run_gemm<DD, DD, 2, false, false, true, false>(eb, Wkb, nullptr, Kef, nullptr, BB * LL, stream);
  run_gemm<DD, DD, 2, false, false, true, false>(eb, Wvb, nullptr, Vef, nullptr, BB * LL, stream);

  for (int cyc = 0; cyc < NCYC; ++cyc) {
    // token-attention projections
    run_gemm<DD, DD, 2, false, false, true, false>(hb, Wqb, nullptr, Qf,  nullptr, BB * LL, stream);
    run_gemm<DD, DD, 2, false, false, true, false>(hb, Wkb, nullptr, Khf, nullptr, BB * LL, stream);
    run_gemm<DD, DD, 2, false, false, true, false>(hb, Wvb, nullptr, Vhf, nullptr, BB * LL, stream);
    run_gemm<DD, DD, 1, false, false, true, false>(sb, Wkb, nullptr, Ksf, nullptr, BB, stream);
    run_gemm<DD, DD, 1, false, false, true, false>(sb, Wvb, nullptr, Vsf, nullptr, BB, stream);
    // 5-key attention per token
    ring_attn<<<BB * LL / 8, 256, 0, stream>>>(Qf, Khf, Kef, Ksf, Vhf, Vef, Vsf, attnb);
    // h = relu(attn @ Wo + bo)
    run_gemm<DD, DD, 2, true, true, true, true>(attnb, Wob, bo, hf, hb, BB * LL, stream);
    // relay: reproject new h, q from old s (Ks/Vs of old s already computed)
    run_gemm<DD, DD, 2, false, false, true, false>(hb, Wkb, nullptr, Khf, nullptr, BB * LL, stream);
    run_gemm<DD, DD, 2, false, false, true, false>(hb, Wvb, nullptr, Vhf, nullptr, BB * LL, stream);
    run_gemm<DD, DD, 1, false, false, true, false>(sb, Wqb, nullptr, Qsf, nullptr, BB, stream);
    relay_attn<<<BB * HH, 256, 0, stream>>>(Qsf, Ksf, Khf, Vsf, Vhf, satb);
    // s = relu(s_attn @ Wo + bo)
    run_gemm<DD, DD, 1, true, true, true, true>(satb, Wob, bo, sf, sb, BB, stream);
  }

  // out1 = LN(e + h); FFN; out2 = LN(out1 + ffn)
  add_ln<true><<<BB * LL / 8, 256, 0, stream>>>(e, hf, g1, be1, out1f, out1b, BB * LL);
  run_gemm<FF, DD, 2, true, true, false, true>(out1b, W1b, b1, nullptr, ffn1b, BB * LL, stream);
  run_gemm<DD, FF, 2, true, false, true, false>(ffn1b, W2b, b2, ffn2f, nullptr, BB * LL, stream);
  add_ln<false><<<BB * LL / 8, 256, 0, stream>>>(out1f, ffn2f, g2, be2, (float*)d_out, nullptr, BB * LL);
  // second output: s
  copy_kernel<<<(BB * DD + 255) / 256, 256, 0, stream>>>(sf, (float*)d_out + ND, BB * DD);
}
